// AttentionGCN_4836133175766
// MI455X (gfx1250) — matrix operand order repaired and checked by reading, not compiled, not run on hardware
//
#include <hip/hip_runtime.h>

#define BN 8192
#define LN 32
#define DN 128
#define HN 8

typedef __attribute__((ext_vector_type(16))) __bf16   v16bf;
typedef __attribute__((ext_vector_type(2)))  __bf16   v2bf;
typedef __attribute__((ext_vector_type(8)))  float    v8f;
typedef __attribute__((ext_vector_type(2)))  float    v2f;
typedef __attribute__((ext_vector_type(4)))  float    v4f;
typedef __attribute__((ext_vector_type(8)))  unsigned int v8u;
typedef __attribute__((ext_vector_type(4)))  unsigned int v4u;

// ---------------- bf16 helpers ----------------
// Packed pair conversion: lowers to v_cvt_pk_bf16_f32 (RTNE) on gfx1250.
__device__ __forceinline__ unsigned pack2(float f0, float f1) {
  v2f f; f[0] = f0; f[1] = f1;
  v2bf b = __builtin_convertvector(f, v2bf);
  return __builtin_bit_cast(unsigned, b);
}
__device__ __forceinline__ __bf16 f2bf(float f) {
  return (__bf16)f;
}

// ---------------- WMMA wrapper ----------------
__device__ __forceinline__ v8f wmma_bf16(v16bf a, v16bf b, v8f c) {
  return __builtin_amdgcn_wmma_f32_16x16x32_bf16(false, a, false, b, (short)0, c,
                                                 false, false);
}

__device__ __forceinline__ v16bf frag_from(v4u lo, v4u hi) {
  v8u t = __builtin_shufflevector(lo, hi, 0, 1, 2, 3, 4, 5, 6, 7);
  return __builtin_bit_cast(v16bf, t);
}

// A fragment 16x32 bf16 from row-major [M>=16][stride] bf16 source.
// lane: M = ln&15 ; K = kt*32 + (ln>>4)*16 + {0..15}.
__device__ __forceinline__ v16bf load_a(const __bf16* base, int stride, int kt, int ln) {
  int row = ln & 15;
  int kh  = (ln >> 4) << 4;
  const __bf16* p = base + row * stride + kt * 32 + kh;
  v4u lo = *(const v4u*)(const void*)(p);
  v4u hi = *(const v4u*)(const void*)(p + 8);
  return frag_from(lo, hi);
}

// A fragment for q viewed as [heads 16(pad)][128]; pad rows replicate h&7.
__device__ __forceinline__ v16bf load_a_q(const __bf16* qrow, int kt, int ln) {
  int h  = ln & 7;
  int kh = (ln >> 4) << 4;
  const __bf16* p = qrow + h * DN + kt * 32 + kh;
  v4u lo = *(const v4u*)(const void*)(p);
  v4u hi = *(const v4u*)(const void*)(p + 8);
  return frag_from(lo, hi);
}

// B fragment 32x16 from [N][K] row-major bf16 (weights): lane reads 16 consecutive K.
__device__ __forceinline__ v16bf load_b_nk(const __bf16* base, int strideK, int ntile,
                                           int kt, int ln) {
  int n  = ntile * 16 + (ln & 15);
  int kb = kt * 32 + ((ln >> 4) << 4);
  const __bf16* p = base + (size_t)n * strideK + kb;
  v4u lo = *(const v4u*)(const void*)(p);
  v4u hi = *(const v4u*)(const void*)(p + 8);
  return frag_from(lo, hi);
}

// B fragment from [N][K] row-major fp32 (neigh^T for scores), converted to bf16.
__device__ __forceinline__ v16bf load_b_nk_f32(const float* base, int strideK, int ntile,
                                               int kt, int ln) {
  int n  = ntile * 16 + (ln & 15);
  int kb = kt * 32 + ((ln >> 4) << 4);
  const v4f* q = (const v4f*)(const void*)(base + (size_t)n * strideK + kb);
  v4f f0 = q[0], f1 = q[1], f2 = q[2], f3 = q[3];
  v8u t;
  t[0] = pack2(f0[0], f0[1]); t[1] = pack2(f0[2], f0[3]);
  t[2] = pack2(f1[0], f1[1]); t[3] = pack2(f1[2], f1[3]);
  t[4] = pack2(f2[0], f2[1]); t[5] = pack2(f2[2], f2[3]);
  t[6] = pack2(f3[0], f3[1]); t[7] = pack2(f3[2], f3[3]);
  return __builtin_bit_cast(v16bf, t);
}

// B fragment from [K][N] row-major fp32 (neigh for context GEMM), K=32 single tile.
__device__ __forceinline__ v16bf load_b_kn_f32(const float* base, int strideN, int ntile,
                                               int ln) {
  int n  = ntile * 16 + (ln & 15);
  int kb = (ln >> 4) << 4;
  v8u t;
#pragma unroll
  for (int r = 0; r < 8; ++r) {
    float f0 = base[(size_t)(kb + 2 * r)     * strideN + n];
    float f1 = base[(size_t)(kb + 2 * r + 1) * strideN + n];
    t[r] = pack2(f0, f1);
  }
  return __builtin_bit_cast(v16bf, t);
}

// ---------------- fused attention layer (16 nodes / workgroup) ----------------
template <bool LAST>
__device__ void attn_layer(__bf16* sX, __bf16* sQC, __bf16* sP,
                           const float* __restrict__ gNeighTile,
                           const int* __restrict__ gDegTile,
                           const __bf16* __restrict__ wqb, const float* __restrict__ bq,
                           const __bf16* __restrict__ Wp,  const __bf16* __restrict__ wpb,
                           const float* __restrict__ cbias,
                           float* __restrict__ gOutTile) {
  const int tid = threadIdx.x;
  const int ln  = tid & 31;
  const int w   = tid >> 5;  // wave 0..15
  const float scale = 0.088388347648318447f;  // 1/sqrt(128)

  // ---- Phase Q: q = x @ wq^T + bq  -> sQC [16 nodes][1024] bf16
#pragma unroll
  for (int t = 0; t < 4; ++t) {
    int nt = w * 4 + t;
    v8f acc = {};
#pragma unroll
    for (int kt = 0; kt < 4; ++kt) {
      v16bf a = load_a(sX, DN, kt, ln);
      v16bf b = load_b_nk(wqb, DN, nt, kt, ln);
      acc = wmma_bf16(a, b, acc);
    }
    float bias = bq[nt * 16 + (ln & 15)];
    int col = nt * 16 + (ln & 15);
    int mb  = (ln >> 4) * 8;
#pragma unroll
    for (int v = 0; v < 8; ++v)
      sQC[(size_t)(mb + v) * (HN * DN) + col] = f2bf(acc[v] + bias);
  }
  __syncthreads();

  // ---- Phase S: per-node scores [16x32] + masked softmax -> sP (bf16, rows 8..15 zero)
  {
    int m = w;
    const float* nb = gNeighTile + (size_t)m * LN * DN;
    int deg = gDegTile[m];
    v8f a0 = {}, a1 = {};
#pragma unroll
    for (int kt = 0; kt < 4; ++kt) {
      v16bf a  = load_a_q(sQC + (size_t)m * (HN * DN), kt, ln);
      v16bf b0 = load_b_nk_f32(nb, DN, 0, kt, ln);
      a0 = wmma_bf16(a, b0, a0);
      v16bf b1 = load_b_nk_f32(nb, DN, 1, kt, ln);
      a1 = wmma_bf16(a, b1, a1);
    }
    int l0 = ln & 15, l1 = 16 + (ln & 15);
    float p0[8], p1[8];
#pragma unroll
    for (int v = 0; v < 8; ++v) {
      float s0 = (l0 < deg) ? a0[v] * scale : -1e30f;
      float s1 = (l1 < deg) ? a1[v] * scale : -1e30f;
      float mx = fmaxf(s0, s1);
#pragma unroll
      for (int off = 8; off >= 1; off >>= 1) mx = fmaxf(mx, __shfl_xor(mx, off, 32));
      float e0 = __expf(s0 - mx), e1 = __expf(s1 - mx);
      float ts = e0 + e1;
#pragma unroll
      for (int off = 8; off >= 1; off >>= 1) ts += __shfl_xor(ts, off, 32);
      float r = 1.0f / ts;
      p0[v] = e0 * r; p1[v] = e1 * r;
    }
    if (ln < 16) {
#pragma unroll
      for (int v = 0; v < 8; ++v) {
        sP[m * 16 * LN + v * LN + l0] = f2bf(p0[v]);
        sP[m * 16 * LN + v * LN + l1] = f2bf(p1[v]);
      }
    }
  }
  __syncthreads();

  // ---- Phase C: c[h,:] = p[h,:] @ neigh -> reuse sQC as c_flat [16 nodes][1024]
  {
    int m = w;
    const float* nb = gNeighTile + (size_t)m * LN * DN;
    v16bf a = load_a(sP + m * 16 * LN, LN, 0, ln);
#pragma unroll
    for (int nt = 0; nt < 8; ++nt) {
      v16bf b = load_b_kn_f32(nb, DN, nt, ln);
      v8f acc = {};
      acc = wmma_bf16(a, b, acc);
      if (ln < 16) {
        int d = nt * 16 + ln;
#pragma unroll
        for (int v = 0; v < 8; ++v)
          sQC[(size_t)m * (HN * DN) + v * DN + d] = f2bf(acc[v]);
      }
    }
  }
  __syncthreads();

  // ---- Phase O: out = c_flat @ Wp + x @ wp^T + (bp + bv-fold)   (waves 0..7)
  v8f acc = {};
  if (w < 8) {
    int nt = w;
#pragma unroll
    for (int kt = 0; kt < 32; ++kt) {
      v16bf a = load_a(sQC, HN * DN, kt, ln);
      v16bf b = load_b_nk(Wp, HN * DN, nt, kt, ln);
      acc = wmma_bf16(a, b, acc);
    }
#pragma unroll
    for (int kt = 0; kt < 4; ++kt) {
      v16bf a = load_a(sX, DN, kt, ln);
      v16bf b = load_b_nk(wpb, DN, nt, kt, ln);
      acc = wmma_bf16(a, b, acc);
    }
    float cb = cbias[nt * 16 + (ln & 15)];
#pragma unroll
    for (int v = 0; v < 8; ++v) acc[v] += cb;
  }
  __syncthreads();  // everyone done reading sX before rewrite
  if (w < 8) {
    int nt  = w;
    int col = nt * 16 + (ln & 15);
    int mb  = (ln >> 4) * 8;
    if (LAST) {
#pragma unroll
      for (int v = 0; v < 8; ++v)
        gOutTile[(size_t)(mb + v) * DN + col] = acc[v];
    } else {
#pragma unroll
      for (int v = 0; v < 8; ++v)
        sX[(mb + v) * DN + col] = f2bf(fmaxf(acc[v], 0.0f));
    }
  }
  __syncthreads();
}

// ---------------- main kernel ----------------
__global__ __launch_bounds__(512) void attn_gcn_kernel(
    const float* __restrict__ gNode, const float* __restrict__ gNeigh,
    const int* __restrict__ gDeg,
    const __bf16* __restrict__ wq0b, const float* __restrict__ bq0,
    const __bf16* __restrict__ wq1b, const float* __restrict__ bq1,
    const __bf16* __restrict__ W0p,  const __bf16* __restrict__ W1p,
    const __bf16* __restrict__ wp0b, const __bf16* __restrict__ wp1b,
    const float* __restrict__ cb0,   const float* __restrict__ cb1,
    float* __restrict__ gOut) {
  __shared__ __align__(16) __bf16 sX[16 * DN];           //  4 KB  (layer input, bf16)
  __shared__ __align__(16) __bf16 sQC[16 * HN * DN];     // 32 KB  (q, then c_flat)
  __shared__ __align__(16) __bf16 sP[16 * 16 * LN];      // 16 KB  (softmax probs, padded)

  const int tid      = threadIdx.x;
  const int tileBase = blockIdx.x * 16;

  // stage node embeddings (bf16) ; vector packed conversion, 2 at a time
  {
    const v2f* src = (const v2f*)(const void*)(gNode + (size_t)tileBase * DN);
    unsigned* dst  = (unsigned*)(void*)sX;
    for (int i = tid; i < (16 * DN) / 2; i += 512) {
      v2f f = src[i];
      dst[i] = pack2(f[0], f[1]);
    }
  }
  // zero sP with wide stores (rows 8..15 stay zero for the whole kernel)
  {
    v4u z = {};
    v4u* dst = (v4u*)(void*)sP;
    for (int i = tid; i < (16 * 16 * LN) / 8; i += 512) dst[i] = z;
  }
  __syncthreads();

  const float* gNeighTile = gNeigh + (size_t)tileBase * LN * DN;
  const int*   gDegTile   = gDeg + tileBase;
  float*       gOutTile   = gOut + (size_t)tileBase * DN;

  attn_layer<false>(sX, sQC, sP, gNeighTile, gDegTile, wq0b, bq0, W0p, wp0b, cb0, nullptr);
  attn_layer<true >(sX, sQC, sP, gNeighTile, gDegTile, wq1b, bq1, W1p, wp1b, cb1, gOutTile);
}

// ---------------- weight prep: bf16 convert + fold value-proj / biases ----------------
__global__ void prep_kernel(
    const float* __restrict__ wq0, const float* __restrict__ wv0,
    const float* __restrict__ bv0, const float* __restrict__ wp0,
    const float* __restrict__ bp0, const float* __restrict__ wq1,
    const float* __restrict__ wv1, const float* __restrict__ bv1,
    const float* __restrict__ wp1, const float* __restrict__ bp1,
    __bf16* wq0b, __bf16* wq1b, __bf16* wp0b, __bf16* wp1b,
    __bf16* W0p, __bf16* W1p, float* cb0, float* cb1) {
  int i = blockIdx.x * blockDim.x + threadIdx.x;
  if (i < 1024 * DN) { wq0b[i] = f2bf(wq0[i]); wq1b[i] = f2bf(wq1[i]); }
  if (i < DN * DN)   { wp0b[i] = f2bf(wp0[i]); wp1b[i] = f2bf(wp1[i]); }
  if (i < DN * 1024) {
    int n  = i >> 10;      // output feature (row), [N][K] layout, K=1024
    int k  = i & 1023;
    int h0 = n >> 4;       // layer0 block-diag head (n = h*16 + jj)
    int g  = k >> 7;       // K-block head
    int kk = k & 127;
    W0p[i] = (g == h0) ? f2bf(wv0[n * DN + kk]) : f2bf(0.0f);
    W1p[i] = f2bf(wv1[(size_t)(g * DN + n) * DN + kk] * 0.125f);  // head-mean fold
  }
  if (i < DN) {
    cb0[i] = bp0[i] + bv0[i];  // softmax rows sum to 1 -> bv passes through
    float s = 0.0f;
    for (int h = 0; h < HN; ++h) s += bv1[h * DN + i];
    cb1[i] = bp1[i] + 0.125f * s;
  }
}

extern "C" void kernel_launch(void* const* d_in, const int* in_sizes, int n_in,
                              void* d_out, int out_size, void* d_ws, size_t ws_size,
                              hipStream_t stream) {
  const float* node  = (const float*)d_in[0];
  const float* neigh = (const float*)d_in[1];
  const int*   deg   = (const int*)d_in[2];
  const float* wq0 = (const float*)d_in[3];
  const float* bq0 = (const float*)d_in[4];
  const float* wv0 = (const float*)d_in[5];
  const float* bv0 = (const float*)d_in[6];
  const float* wp0 = (const float*)d_in[7];
  const float* bp0 = (const float*)d_in[8];
  const float* wq1 = (const float*)d_in[9];
  const float* bq1 = (const float*)d_in[10];
  const float* wv1 = (const float*)d_in[11];
  const float* bv1 = (const float*)d_in[12];
  const float* wp1 = (const float*)d_in[13];
  const float* bp1 = (const float*)d_in[14];

  char* ws = (char*)d_ws;
  __bf16* wq0b = (__bf16*)(ws + 0);        // 1024*128 bf16 = 256 KB
  __bf16* wq1b = (__bf16*)(ws + 262144);   // 256 KB
  __bf16* wp0b = (__bf16*)(ws + 524288);   // 32 KB
  __bf16* wp1b = (__bf16*)(ws + 557056);   // 32 KB
  __bf16* W0p  = (__bf16*)(ws + 589824);   // 256 KB  (block-diag folded wv0)
  __bf16* W1p  = (__bf16*)(ws + 851968);   // 256 KB  (head-mean folded wv1)
  float*  cb0  = (float*)(ws + 1114112);   // 512 B
  float*  cb1  = (float*)(ws + 1114624);   // 512 B

  prep_kernel<<<512, 256, 0, stream>>>(wq0, wv0, bv0, wp0, bp0,
                                       wq1, wv1, bv1, wp1, bp1,
                                       wq0b, wq1b, wp0b, wp1b, W0p, W1p, cb0, cb1);
  attn_gcn_kernel<<<BN / 16, 512, 0, stream>>>(node, neigh, deg,
                                               wq0b, bq0, wq1b, bq1,
                                               W0p, W1p, wp0b, wp1b,
                                               cb0, cb1, (float*)d_out);
}
